// SparseMultiHeadAttention_15255723836127
// MI455X (gfx1250) — compile-verified, matrix operand order checked by reading, not run on hardware
//
#include <hip/hip_runtime.h>

// ---------------------------------------------------------------------------
// Sparse (windowed + global) multi-head attention for MI455X (gfx1250).
// All matmuls via v_wmma_f32_16x16x32_bf16.
// ---------------------------------------------------------------------------

#define S_LEN   4096
#define D_MODEL 1024
#define N_HEADS 16
#define D_K     64
#define WIN     256
#define GTOK    64

typedef __attribute__((ext_vector_type(16))) __bf16 v16bf;
typedef __attribute__((ext_vector_type(8)))  __bf16 v8bf;
typedef __attribute__((ext_vector_type(8)))  float  v8f;

__device__ __forceinline__ unsigned short f32_to_bf16(float f) {
    unsigned int u = __float_as_uint(f);
    unsigned int lsb = (u >> 16) & 1u;
    u += 0x7fffu + lsb;                 // round to nearest even
    return (unsigned short)(u >> 16);
}

// Load one 16x32 (A) / 32x16 (B) bf16 fragment: 8 contiguous bf16 at p,
// 8 contiguous bf16 at p+16 (ISA 16-bit WMMA operand layout, wave32).
__device__ __forceinline__ v16bf ldfrag(const unsigned short* p) {
    v8bf lo = *reinterpret_cast<const v8bf*>(p);
    v8bf hi = *reinterpret_cast<const v8bf*>(p + 16);
    v16bf r;
#pragma unroll
    for (int i = 0; i < 8; ++i) { r[i] = lo[i]; r[i + 8] = hi[i]; }
    return r;
}

__device__ __forceinline__ v8f wmma_bf16(v16bf a, v16bf b, v8f c) {
    return __builtin_amdgcn_wmma_f32_16x16x32_bf16(
        /*neg_a=*/false, a, /*neg_b=*/false, b,
        /*c_mod=*/(short)0, c, /*reuse_a=*/false, /*reuse_b=*/false);
}

// ---------------------------------------------------------------------------
// f32 -> bf16 conversion
// ---------------------------------------------------------------------------
__global__ void cvt_bf16_kernel(const float* __restrict__ src,
                                unsigned short* __restrict__ dst, int n) {
    int i = blockIdx.x * blockDim.x + threadIdx.x;
    if (i < n) dst[i] = f32_to_bf16(src[i]);
}

// ---------------------------------------------------------------------------
// NT GEMM: C(M x N) = A(M x K) @ W(N x K)^T + bias, bf16 inputs, f32 accum.
// M=4096, N=K=1024. Block = 256 threads (8 waves). Block tile = 128 x 64.
// Each wave computes a 16 x 64 strip with 4 WMMA accumulators.
// MODE 0: f32 row-major out (final projection).
// MODE 1: bf16 out as (H, S, 64)   head-major (Q, K).
// MODE 2: bf16 out as (H, 64, S)   transposed (V^T).
// ---------------------------------------------------------------------------
template <int MODE>
__global__ __launch_bounds__(256) void gemm_bf16_nt(
    const unsigned short* __restrict__ A,
    const unsigned short* __restrict__ Wt,
    const float* __restrict__ bias,
    float* __restrict__ outF,
    unsigned short* __restrict__ outB) {
    const int K = D_MODEL, N = D_MODEL;
    const int lane = threadIdx.x & 31;
    const int wave = threadIdx.x >> 5;
    const int m0 = blockIdx.x * 128 + wave * 16;
    const int n0 = blockIdx.y * 64;

    const int l16  = lane & 15;
    const int koff = (lane >> 4) << 3;   // K offset 0 or 8 (half-wave split)

    const unsigned short* arow = A + (size_t)(m0 + l16) * K + koff;
    const unsigned short* wr0 = Wt + (size_t)(n0 +  0 + l16) * K + koff;
    const unsigned short* wr1 = Wt + (size_t)(n0 + 16 + l16) * K + koff;
    const unsigned short* wr2 = Wt + (size_t)(n0 + 32 + l16) * K + koff;
    const unsigned short* wr3 = Wt + (size_t)(n0 + 48 + l16) * K + koff;

    v8f acc[4] = {v8f{}, v8f{}, v8f{}, v8f{}};

    for (int k = 0; k < K; k += 32) {
        v16bf a  = ldfrag(arow + k);
        v16bf b0 = ldfrag(wr0 + k);
        v16bf b1 = ldfrag(wr1 + k);
        v16bf b2 = ldfrag(wr2 + k);
        v16bf b3 = ldfrag(wr3 + k);
        acc[0] = wmma_bf16(a, b0, acc[0]);
        acc[1] = wmma_bf16(a, b1, acc[1]);
        acc[2] = wmma_bf16(a, b2, acc[2]);
        acc[3] = wmma_bf16(a, b3, acc[3]);
    }

    const int hi8 = (lane >> 4) << 3;    // C layout: rows r+0 (lanes 0-15) / r+8
#pragma unroll
    for (int t = 0; t < 4; ++t) {
        int ncol = n0 + t * 16 + l16;
        float bv = bias[ncol];
#pragma unroll
        for (int r = 0; r < 8; ++r) {
            int mrow = m0 + r + hi8;
            float v = acc[t][r] + bv;
            if (MODE == 0) {
                outF[(size_t)mrow * N + ncol] = v;
            } else if (MODE == 1) {
                int h = ncol >> 6, d = ncol & 63;
                outB[((size_t)h * S_LEN + mrow) * D_K + d] = f32_to_bf16(v);
            } else {
                int h = ncol >> 6, d = ncol & 63;
                outB[((size_t)h * D_K + d) * S_LEN + mrow] = f32_to_bf16(v);
            }
        }
    }
}

// ---------------------------------------------------------------------------
// Flash-style sparse attention. Grid = (H, S/128). 8 waves/block, each wave
// owns one 16-query block and runs an independent online-softmax key loop
// over 32-key chunks restricted to [0,64) U [qr0-256, qr0+15+256].
// Qh,Kh: (H,S,64) bf16. VT: (H,64,S) bf16. Ob: (S, 1024) bf16.
// ---------------------------------------------------------------------------
__global__ __launch_bounds__(256) void sparse_attn_kernel(
    const unsigned short* __restrict__ Qh,
    const unsigned short* __restrict__ Kh,
    const unsigned short* __restrict__ VT,
    unsigned short* __restrict__ Ob) {
    __shared__ unsigned short ldsP[8][16 * 32];   // 1KB per wave P-tile bounce

    const int lane = threadIdx.x & 31;
    const int wave = threadIdx.x >> 5;
    const int h    = blockIdx.x;
    const int qr0  = blockIdx.y * 128 + wave * 16;

    const int l16  = lane & 15;
    const int koff = (lane >> 4) << 3;
    const int hi8  = koff;                       // same value, C-layout row split

    const unsigned short* Qb = Qh + (size_t)h * S_LEN * D_K;
    const unsigned short* Kb = Kh + (size_t)h * S_LEN * D_K;
    const unsigned short* Vb = VT + (size_t)h * D_K * S_LEN;

    // Q fragments (A operand), contraction dim d split 0..31 / 32..63
    const unsigned short* qrow = Qb + (size_t)(qr0 + l16) * D_K + koff;
    v16bf aQ0 = ldfrag(qrow + 0);
    v16bf aQ1 = ldfrag(qrow + 32);

    float m_run[8], l_run[8];
#pragma unroll
    for (int r = 0; r < 8; ++r) { m_run[r] = -1e30f; l_run[r] = 0.0f; }
    v8f o[4] = {v8f{}, v8f{}, v8f{}, v8f{}};

    // Key ranges: prefix [0, pend) for global tokens, window [wstart, wend).
    int wstart, wend;
    if (qr0 + 15 < GTOK) {               // all rows are global queries
        wstart = 0; wend = S_LEN;
    } else {
        int ws = qr0 - WIN; if (ws < 0) ws = 0;
        wstart = ws & ~31;
        int we = qr0 + 15 + WIN + 1; if (we > S_LEN) we = S_LEN;
        wend = (we + 31) & ~31;
    }
    int pend = (wstart < GTOK) ? wstart : GTOK;   // min(GTOK, wstart)

    unsigned short* myP = &ldsP[wave][0];

    for (int seg = 0; seg < 2; ++seg) {
        const int cbeg = seg ? wstart : 0;
        const int cend = seg ? wend : pend;
        for (int c = cbeg; c < cend; c += 32) {
            // ---- S = Q @ K^T for 32 keys (two 16x16 tiles) ----
            const unsigned short* kr0 = Kb + (size_t)(c + l16) * D_K + koff;
            const unsigned short* kr1 = Kb + (size_t)(c + 16 + l16) * D_K + koff;
            v16bf bK00 = ldfrag(kr0 + 0);
            v16bf bK01 = ldfrag(kr0 + 32);
            v16bf bK10 = ldfrag(kr1 + 0);
            v16bf bK11 = ldfrag(kr1 + 32);
            v8f s0 = {}, s1 = {};
            s0 = wmma_bf16(aQ0, bK00, s0);
            s0 = wmma_bf16(aQ1, bK01, s0);
            s1 = wmma_bf16(aQ0, bK10, s1);
            s1 = wmma_bf16(aQ1, bK11, s1);

            // ---- scale + sparsity mask ----
            const int j0 = c + l16;
            const int j1 = c + 16 + l16;
#pragma unroll
            for (int r = 0; r < 8; ++r) {
                int i = qr0 + r + hi8;
                int d0 = i - j0; if (d0 < 0) d0 = -d0;
                int d1 = i - j1; if (d1 < 0) d1 = -d1;
                bool a0 = (i < GTOK) || (j0 < GTOK) || (d0 <= WIN);
                bool a1 = (i < GTOK) || (j1 < GTOK) || (d1 <= WIN);
                s0[r] = a0 ? s0[r] * 0.125f : -1e30f;
                s1[r] = a1 ? s1[r] * 0.125f : -1e30f;
            }

            // ---- online softmax: row max (reduce across 16-lane group) ----
            float mx[8];
#pragma unroll
            for (int r = 0; r < 8; ++r) mx[r] = fmaxf(s0[r], s1[r]);
#pragma unroll
            for (int off = 1; off < 16; off <<= 1) {
#pragma unroll
                for (int r = 0; r < 8; ++r)
                    mx[r] = fmaxf(mx[r], __shfl_xor(mx[r], off, 32));
            }
            float alpha[8];
#pragma unroll
            for (int r = 0; r < 8; ++r) {
                float mn = fmaxf(m_run[r], mx[r]);
                alpha[r] = __expf(m_run[r] - mn);
                m_run[r] = mn;
            }
#pragma unroll
            for (int r = 0; r < 8; ++r) {
                s0[r] = __expf(s0[r] - m_run[r]);
                s1[r] = __expf(s1[r] - m_run[r]);
            }
            float rs[8];
#pragma unroll
            for (int r = 0; r < 8; ++r) rs[r] = s0[r] + s1[r];
#pragma unroll
            for (int off = 1; off < 16; off <<= 1) {
#pragma unroll
                for (int r = 0; r < 8; ++r)
                    rs[r] += __shfl_xor(rs[r], off, 32);
            }
#pragma unroll
            for (int r = 0; r < 8; ++r)
                l_run[r] = l_run[r] * alpha[r] + rs[r];
#pragma unroll
            for (int t = 0; t < 4; ++t)
#pragma unroll
                for (int r = 0; r < 8; ++r) o[t][r] *= alpha[r];

            // ---- relayout P (C layout) -> A operand via LDS bounce ----
#pragma unroll
            for (int r = 0; r < 8; ++r) {
                myP[(r + hi8) * 32 + l16]      = f32_to_bf16(s0[r]);
                myP[(r + hi8) * 32 + 16 + l16] = f32_to_bf16(s1[r]);
            }
            asm volatile("s_wait_dscnt 0" ::: "memory");
            v16bf aP = ldfrag(myP + l16 * 32 + koff);

            // ---- O += P @ V  (V^T rows are key-contiguous) ----
#pragma unroll
            for (int t = 0; t < 4; ++t) {
                const unsigned short* vr =
                    Vb + (size_t)(t * 16 + l16) * S_LEN + c + koff;
                v16bf bV = ldfrag(vr);
                o[t] = wmma_bf16(aP, bV, o[t]);
            }
        }
    }

    // ---- normalize and store as (S, H*64) bf16 for the output projection ----
    float inv[8];
#pragma unroll
    for (int r = 0; r < 8; ++r) inv[r] = 1.0f / l_run[r];
#pragma unroll
    for (int t = 0; t < 4; ++t) {
        int col = h * D_K + t * 16 + l16;
#pragma unroll
        for (int r = 0; r < 8; ++r) {
            int srow = qr0 + r + hi8;
            Ob[(size_t)srow * D_MODEL + col] = f32_to_bf16(o[t][r] * inv[r]);
        }
    }
}

// ---------------------------------------------------------------------------
// Launch
// ---------------------------------------------------------------------------
extern "C" void kernel_launch(void* const* d_in, const int* in_sizes, int n_in,
                              void* d_out, int out_size, void* d_ws, size_t ws_size,
                              hipStream_t stream) {
    const float* q   = (const float*)d_in[0];
    const float* k   = (const float*)d_in[1];
    const float* v   = (const float*)d_in[2];
    const float* w_q = (const float*)d_in[3];
    const float* b_q = (const float*)d_in[4];
    const float* w_k = (const float*)d_in[5];
    const float* b_k = (const float*)d_in[6];
    const float* w_v = (const float*)d_in[7];
    const float* b_v = (const float*)d_in[8];
    const float* w_o = (const float*)d_in[9];
    const float* b_o = (const float*)d_in[10];

    const size_t ACT = (size_t)S_LEN * D_MODEL;   // 4M elements
    const size_t WGT = (size_t)D_MODEL * D_MODEL; // 1M elements

    unsigned short* p = (unsigned short*)d_ws;
    unsigned short* qbf  = p; p += ACT;
    unsigned short* kbf  = p; p += ACT;
    unsigned short* vbf  = p; p += ACT;
    unsigned short* wqbf = p; p += WGT;
    unsigned short* wkbf = p; p += WGT;
    unsigned short* wvbf = p; p += WGT;
    unsigned short* wobf = p; p += WGT;
    unsigned short* Qh   = p; p += ACT;   // (H,S,64)
    unsigned short* Kh   = p; p += ACT;   // (H,S,64)
    unsigned short* VTh  = p; p += ACT;   // (H,64,S)
    unsigned short* Obf  = p; p += ACT;   // (S, 1024)

    const int TPB = 256;
    cvt_bf16_kernel<<<(int)((ACT + TPB - 1) / TPB), TPB, 0, stream>>>(q, qbf, (int)ACT);
    cvt_bf16_kernel<<<(int)((ACT + TPB - 1) / TPB), TPB, 0, stream>>>(k, kbf, (int)ACT);
    cvt_bf16_kernel<<<(int)((ACT + TPB - 1) / TPB), TPB, 0, stream>>>(v, vbf, (int)ACT);
    cvt_bf16_kernel<<<(int)((WGT + TPB - 1) / TPB), TPB, 0, stream>>>(w_q, wqbf, (int)WGT);
    cvt_bf16_kernel<<<(int)((WGT + TPB - 1) / TPB), TPB, 0, stream>>>(w_k, wkbf, (int)WGT);
    cvt_bf16_kernel<<<(int)((WGT + TPB - 1) / TPB), TPB, 0, stream>>>(w_v, wvbf, (int)WGT);
    cvt_bf16_kernel<<<(int)((WGT + TPB - 1) / TPB), TPB, 0, stream>>>(w_o, wobf, (int)WGT);

    dim3 gg(S_LEN / 128, D_MODEL / 64);   // (32, 16)
    gemm_bf16_nt<1><<<gg, TPB, 0, stream>>>(qbf, wqbf, b_q, nullptr, Qh);
    gemm_bf16_nt<1><<<gg, TPB, 0, stream>>>(kbf, wkbf, b_k, nullptr, Kh);
    gemm_bf16_nt<2><<<gg, TPB, 0, stream>>>(vbf, wvbf, b_v, nullptr, VTh);

    dim3 ga(N_HEADS, S_LEN / 128);        // (16, 32)
    sparse_attn_kernel<<<ga, TPB, 0, stream>>>(Qh, Kh, VTh, Obf);

    gemm_bf16_nt<0><<<gg, TPB, 0, stream>>>(Obf, wobf, b_o, (float*)d_out, nullptr);
}
